// MoDBlock_43748536877278
// MI455X (gfx1250) — matrix operand order repaired and checked by reading, not compiled, not run on hardware
//
#include <hip/hip_runtime.h>

typedef __attribute__((ext_vector_type(16))) __bf16 v16bf;
typedef __attribute__((ext_vector_type(8)))  float  v8f;
typedef __attribute__((ext_vector_type(4)))  int    v4i;

#define DMODEL 1024
#define NHEADS 16
#define SEQ    4096
#define BATCH  4
#define KTOK   512
#define ROWS   (BATCH*KTOK)   // 2048

#if defined(__has_builtin)
#if __has_builtin(__builtin_amdgcn_global_load_async_to_lds_b128)
#define HAVE_ASYNC 1
#else
#define HAVE_ASYNC 0
#endif
#else
#define HAVE_ASYNC 0
#endif

#define AS1 __attribute__((address_space(1)))
#define AS3 __attribute__((address_space(3)))

// 16-byte global -> LDS copy; async (ASYNCcnt-tracked) path on gfx1250 when available.
__device__ __forceinline__ void async_copy16(const unsigned short* g, unsigned short* l) {
#if HAVE_ASYNC
    __builtin_amdgcn_global_load_async_to_lds_b128((AS1 v4i*)g, (AS3 v4i*)l, 0, 0);
#else
    *(uint4*)l = *(const uint4*)g;
#endif
}

__device__ __forceinline__ void async_wait() {
#if HAVE_ASYNC
#if __has_builtin(__builtin_amdgcn_s_wait_asynccnt)
    __builtin_amdgcn_s_wait_asynccnt(0);
#else
    asm volatile("s_wait_asynccnt 0" ::: "memory");
#endif
#endif
}

__device__ __forceinline__ unsigned short f2bf(float f) {
    unsigned int u = __float_as_uint(f);
    u += 0x7FFFu + ((u >> 16) & 1u);          // round-to-nearest-even
    return (unsigned short)(u >> 16);
}

__device__ __forceinline__ float gelu_tanh(float x) {
    float x3 = x * x * x;
    return 0.5f * x * (1.0f + tanhf(0.7978845608028654f * (x + 0.044715f * x3)));
}

union FragU { v16bf v; uint4 u[2]; };

// A-matrix 16x32 bf16 fragment (ISA layout): lane<16 holds K{kh*8..+7, 16+kh*8..+7}, kh=lane>>4
__device__ __forceinline__ v16bf load_a_frag(const unsigned short* p0, int stride, int m, int lane) {
    FragU f;
    const unsigned short* p = p0 + m * stride + ((lane >> 4) * 8);
    f.u[0] = *(const uint4*)(p);
    f.u[1] = *(const uint4*)(p + 16);
    return f.v;
}

// B-matrix 32x16 bf16 fragment from LDS stored [N][K]: lane holds K{kh*8..+7, 16+kh*8..+7}, kh=lane>>4
__device__ __forceinline__ v16bf load_b_frag(const unsigned short* p0, int stride, int n, int lane) {
    FragU f;
    const unsigned short* p = p0 + n * stride + ((lane >> 4) * 8);
    f.u[0] = *(const uint4*)(p);
    f.u[1] = *(const uint4*)(p + 16);
    return f.v;
}

__device__ __forceinline__ v8f wmma_bf16(v16bf a, v16bf b, v8f c) {
    return __builtin_amdgcn_wmma_f32_16x16x32_bf16(false, a, false, b, (short)0, c, false, false);
}

// ---------------- router: tw[b,s] = x[b,s,:] . w_router + b_router ----------------
__global__ __launch_bounds__(256) void router_kernel(const float* __restrict__ x,
                                                     const float* __restrict__ wr,
                                                     const float* __restrict__ br,
                                                     float* __restrict__ tw) {
    int w = threadIdx.x >> 5, lane = threadIdx.x & 31;
    int tok = blockIdx.x * 8 + w;
    const float* p = x + (size_t)tok * DMODEL;
    float s = 0.f;
#pragma unroll 8
    for (int i = 0; i < 32; ++i) {
        int d = lane + 32 * i;
        s += p[d] * wr[d];
    }
    for (int m = 16; m >= 1; m >>= 1) s += __shfl_xor(s, m, 32);
    if (lane == 0) tw[tok] = s + br[0];
}

// ---------------- top-k by rank counting + ascending compaction ----------------
__global__ __launch_bounds__(1024) void topk_kernel(const float* __restrict__ tw, int* __restrict__ pos) {
    __shared__ float wsh[SEQ];
    __shared__ int scan[1024];
    int b = blockIdx.x, t = threadIdx.x;
    for (int i = t; i < SEQ; i += 1024) wsh[i] = tw[b * SEQ + i];
    __syncthreads();
    float w0 = wsh[4 * t], w1 = wsh[4 * t + 1], w2 = wsh[4 * t + 2], w3 = wsh[4 * t + 3];
    int c0 = 0, c1 = 0, c2 = 0, c3 = 0;
    for (int j = 0; j < SEQ; ++j) {
        float wj = wsh[j];
        c0 += (wj > w0) || (wj == w0 && j < 4 * t);
        c1 += (wj > w1) || (wj == w1 && j < 4 * t + 1);
        c2 += (wj > w2) || (wj == w2 && j < 4 * t + 2);
        c3 += (wj > w3) || (wj == w3 && j < 4 * t + 3);
    }
    int s0 = c0 < KTOK, s1 = c1 < KTOK, s2 = c2 < KTOK, s3 = c3 < KTOK;
    int local = s0 + s1 + s2 + s3;
    scan[t] = local;
    __syncthreads();
    for (int off = 1; off < 1024; off <<= 1) {
        int v = (t >= off) ? scan[t - off] : 0;
        __syncthreads();
        scan[t] += v;
        __syncthreads();
    }
    int o = scan[t] - local + b * KTOK;
    if (s0) pos[o++] = 4 * t;
    if (s1) pos[o++] = 4 * t + 1;
    if (s2) pos[o++] = 4 * t + 2;
    if (s3) pos[o++] = 4 * t + 3;
}

// ---------------- fp32 [K][N] -> bf16 [N][K] transposed weight conversion ----------------
__global__ __launch_bounds__(256) void convert_transpose_kernel(const float* __restrict__ src,
                                                                unsigned short* __restrict__ dst,
                                                                int K, int N) {
    __shared__ unsigned short tile[64][65];
    int k0 = blockIdx.y * 64, n0 = blockIdx.x * 64;
    int t = threadIdx.x;
    for (int idx = t; idx < 4096; idx += 256) {
        int r = idx >> 6, c = idx & 63;
        tile[r][c] = f2bf(src[(size_t)(k0 + r) * N + n0 + c]);
    }
    __syncthreads();
    for (int idx = t; idx < 4096; idx += 256) {
        int n = idx >> 6, k = idx & 63;
        dst[(size_t)(n0 + n) * K + k0 + k] = tile[k][n];
    }
}

// ---------------- gather selected tokens, convert to bf16 ----------------
__global__ __launch_bounds__(256) void gather_kernel(const float* __restrict__ x, const int* __restrict__ pos,
                                                     unsigned short* __restrict__ tok) {
    int i = blockIdx.x;                 // 0..ROWS-1
    int b = i >> 9;
    int p = pos[i];
    const float* src = x + ((size_t)(b * SEQ + p)) * DMODEL;
    unsigned short* dst = tok + (size_t)i * DMODEL;
    for (int d = threadIdx.x; d < DMODEL; d += 256) dst[d] = f2bf(src[d]);
}

// ---------------- tiled bf16 WMMA GEMM: C[M,N] = A[M,K] @ Bt[N,K]^T + bias ----------------
// 128x64 block tile, 8 waves each owning 32x32, double-buffered async LDS staging.
// EPI: 0 = bias + store bf16; 1 = bias + store f32; 2 = bias + gelu + store bf16
template <int EPI>
__global__ __launch_bounds__(256) void gemm_bf16_kernel(const unsigned short* __restrict__ A,
                                                        const unsigned short* __restrict__ Bt,
                                                        const float* __restrict__ bias,
                                                        void* __restrict__ Cout,
                                                        int M, int N, int Kd) {
    __shared__ unsigned short As[2][128][40];   // [m][k], padded
    __shared__ unsigned short Bs[2][64][40];    // [n][k], padded
    const int t = threadIdx.x, lane = t & 31, w = t >> 5;
    const int wm = w & 3, wn = w >> 2;          // wave tile: rows wm*32.., cols wn*32..
    const int rowBase = blockIdx.y * 128, colBase = blockIdx.x * 64;
    v8f acc00 = {}, acc01 = {}, acc10 = {}, acc11 = {};
    const int nk = Kd >> 5;

    auto stage_loads = [&](int kt, int st) {
        // A tile 128x32 = 512 x 16B; 2 per thread
#pragma unroll
        for (int q = 0; q < 2; ++q) {
            int idx = t + q * 256;
            int r = idx >> 2, c0 = (idx & 3) * 8;
            async_copy16(A + (size_t)(rowBase + r) * Kd + kt * 32 + c0, &As[st][r][c0]);
        }
        // B tile 64x32 = 256 x 16B; 1 per thread
        int r = t >> 2, c0 = (t & 3) * 8;
        async_copy16(Bt + (size_t)(colBase + r) * Kd + kt * 32 + c0, &Bs[st][r][c0]);
    };

    stage_loads(0, 0);
    async_wait();
    __syncthreads();
    int cur = 0;
    for (int kt = 0; kt < nk; ++kt) {
        if (kt + 1 < nk) stage_loads(kt + 1, cur ^ 1);   // overlap next tile with compute
        v16bf a0 = load_a_frag(&As[cur][0][0], 40, wm * 32 + (lane & 15), lane);
        v16bf a1 = load_a_frag(&As[cur][0][0], 40, wm * 32 + 16 + (lane & 15), lane);
        v16bf b0 = load_b_frag(&Bs[cur][0][0], 40, wn * 32 + (lane & 15), lane);
        v16bf b1 = load_b_frag(&Bs[cur][0][0], 40, wn * 32 + 16 + (lane & 15), lane);
        acc00 = wmma_bf16(a0, b0, acc00);
        acc01 = wmma_bf16(a0, b1, acc01);
        acc10 = wmma_bf16(a1, b0, acc10);
        acc11 = wmma_bf16(a1, b1, acc11);
        async_wait();
        __syncthreads();
        cur ^= 1;
    }
#pragma unroll
    for (int i = 0; i < 8; ++i) {
        int r0 = rowBase + wm * 32 + (lane >> 4) * 8 + i;
        int r1 = r0 + 16;
        int c0 = colBase + wn * 32 + (lane & 15);
        int c1 = c0 + 16;
        float v00 = acc00[i] + bias[c0];
        float v01 = acc01[i] + bias[c1];
        float v10 = acc10[i] + bias[c0];
        float v11 = acc11[i] + bias[c1];
        if (EPI == 2) {
            v00 = gelu_tanh(v00); v01 = gelu_tanh(v01);
            v10 = gelu_tanh(v10); v11 = gelu_tanh(v11);
        }
        if (EPI == 1) {
            ((float*)Cout)[(size_t)r0 * N + c0] = v00;
            ((float*)Cout)[(size_t)r0 * N + c1] = v01;
            ((float*)Cout)[(size_t)r1 * N + c0] = v10;
            ((float*)Cout)[(size_t)r1 * N + c1] = v11;
        } else {
            ((unsigned short*)Cout)[(size_t)r0 * N + c0] = f2bf(v00);
            ((unsigned short*)Cout)[(size_t)r0 * N + c1] = f2bf(v01);
            ((unsigned short*)Cout)[(size_t)r1 * N + c0] = f2bf(v10);
            ((unsigned short*)Cout)[(size_t)r1 * N + c1] = f2bf(v11);
        }
    }
}

// ---------------- causal flash attention over compacted tokens (bf16 WMMA) ----------------
__global__ __launch_bounds__(256) void attention_kernel(const unsigned short* __restrict__ qkv,
                                                        unsigned short* __restrict__ attn) {
    __shared__ unsigned short Qs[64][72];   // [q][d]
    __shared__ unsigned short Ks[64][72];   // [kv][d]  (== B-layout [N][K] for Q.K^T)
    __shared__ unsigned short Vt[64][72];   // [d][kv]  (== B-layout [N][K] for P.V)
    __shared__ unsigned short Ps[64][72];   // [q][kv]
    __shared__ float Sraw[64][65];
    __shared__ float mrow[64], lrow[64], arow[64];

    const int t = threadIdx.x, lane = t & 31, w = t >> 5;
    const int wm = w & 3, wn = w >> 2;
    const int qb = blockIdx.x & 7;
    const int h  = (blockIdx.x >> 3) & 15;
    const int b  = blockIdx.x >> 7;

    {   // Q tile
        int r = t >> 2, d0 = (t & 3) * 16;
        const unsigned short* src = qkv + ((size_t)(b * KTOK + qb * 64 + r)) * 3072 + h * 64 + d0;
        async_copy16(src, &Qs[r][d0]);
        async_copy16(src + 8, &Qs[r][d0 + 8]);
    }
    if (t < 64) { mrow[t] = -3.0e38f; lrow[t] = 0.f; }
    v8f o0 = {}, o1 = {};
    async_wait();
    __syncthreads();

    for (int j = 0; j <= qb; ++j) {
        {   // K tile (rows are exactly the [N][K] B layout)
            int r = t >> 2, d0 = (t & 3) * 16;
            const unsigned short* src = qkv + ((size_t)(b * KTOK + j * 64 + r)) * 3072 + 1024 + h * 64 + d0;
            async_copy16(src, &Ks[r][d0]);
            async_copy16(src + 8, &Ks[r][d0 + 8]);
        }
        {   // V tile, transposed into Vt[d][kv]
            int r = t >> 2, d0 = (t & 3) * 16;
            union { uint4 q[2]; unsigned short s[16]; } v;
            const unsigned short* src = qkv + ((size_t)(b * KTOK + j * 64 + r)) * 3072 + 2048 + h * 64 + d0;
            v.q[0] = *(const uint4*)(src);
            v.q[1] = *(const uint4*)(src + 8);
#pragma unroll
            for (int jj = 0; jj < 16; ++jj) Vt[d0 + jj][r] = v.s[jj];
        }
        async_wait();
        __syncthreads();

        // phase 1: S = Q @ K^T  (head dim 64 -> 2 WMMA k-steps)
        v8f s0 = {}, s1 = {};
#pragma unroll
        for (int ks = 0; ks < 2; ++ks) {
            v16bf aq  = load_a_frag(&Qs[0][ks * 32], 72, wm * 16 + (lane & 15), lane);
            v16bf bk0 = load_b_frag(&Ks[0][ks * 32], 72, wn * 32 + (lane & 15), lane);
            v16bf bk1 = load_b_frag(&Ks[0][ks * 32], 72, wn * 32 + 16 + (lane & 15), lane);
            s0 = wmma_bf16(aq, bk0, s0);
            s1 = wmma_bf16(aq, bk1, s1);
        }
#pragma unroll
        for (int i = 0; i < 8; ++i) {
            int rl = wm * 16 + (lane >> 4) * 8 + i;
            int cl0 = wn * 32 + (lane & 15), cl1 = cl0 + 16;
            int qg = qb * 64 + rl;
            float v0 = s0[i] * 0.125f, v1 = s1[i] * 0.125f;   // 1/sqrt(64)
            if (j * 64 + cl0 > qg) v0 = -1e9f;
            if (j * 64 + cl1 > qg) v1 = -1e9f;
            Sraw[rl][cl0] = v0;
            Sraw[rl][cl1] = v1;
        }
        __syncthreads();

        // online softmax (one thread per q row)
        if (t < 64) {
            int row = t;
            float mold = mrow[row], mx = mold;
#pragma unroll 8
            for (int c = 0; c < 64; ++c) mx = fmaxf(mx, Sraw[row][c]);
            float alpha = __expf(mold - mx);
            float sum = 0.f;
#pragma unroll 8
            for (int c = 0; c < 64; ++c) {
                float p = __expf(Sraw[row][c] - mx);
                sum += p;
                Ps[row][c] = f2bf(p);
            }
            mrow[row] = mx;
            lrow[row] = lrow[row] * alpha + sum;
            arow[row] = alpha;
        }
        __syncthreads();

        // phase 2: O = O*alpha + P @ V
#pragma unroll
        for (int i = 0; i < 8; ++i) {
            int rl = wm * 16 + (lane >> 4) * 8 + i;
            float a = arow[rl];
            o0[i] *= a;
            o1[i] *= a;
        }
#pragma unroll
        for (int ks = 0; ks < 2; ++ks) {
            v16bf ap  = load_a_frag(&Ps[0][ks * 32], 72, wm * 16 + (lane & 15), lane);
            v16bf bv0 = load_b_frag(&Vt[0][ks * 32], 72, wn * 32 + (lane & 15), lane);
            v16bf bv1 = load_b_frag(&Vt[0][ks * 32], 72, wn * 32 + 16 + (lane & 15), lane);
            o0 = wmma_bf16(ap, bv0, o0);
            o1 = wmma_bf16(ap, bv1, o1);
        }
        __syncthreads();
    }

#pragma unroll
    for (int i = 0; i < 8; ++i) {
        int rl = wm * 16 + (lane >> 4) * 8 + i;
        float inv = 1.0f / lrow[rl];
        int cl0 = wn * 32 + (lane & 15), cl1 = cl0 + 16;
        size_t rowg = (size_t)(b * KTOK + qb * 64 + rl);
        attn[rowg * 1024 + h * 64 + cl0] = f2bf(o0[i] * inv);
        attn[rowg * 1024 + h * 64 + cl1] = f2bf(o1[i] * inv);
    }
}

// ---------------- layernorm (row of 1024 fp32) -> bf16 ----------------
__global__ __launch_bounds__(256) void layernorm_kernel(const float* __restrict__ in,
                                                        const float* __restrict__ g,
                                                        const float* __restrict__ bt,
                                                        unsigned short* __restrict__ out) {
    __shared__ float red[16];
    int row = blockIdx.x, t = threadIdx.x;
    const float* p = in + (size_t)row * DMODEL;
    float s = 0.f, ss = 0.f, vals[4];
#pragma unroll
    for (int i = 0; i < 4; ++i) {
        float v = p[t + 256 * i];
        vals[i] = v; s += v; ss += v * v;
    }
    for (int m = 16; m >= 1; m >>= 1) { s += __shfl_xor(s, m, 32); ss += __shfl_xor(ss, m, 32); }
    int wv = t >> 5, lane = t & 31;
    if (lane == 0) { red[wv] = s; red[8 + wv] = ss; }
    __syncthreads();
    if (t == 0) {
        float S = 0.f, SS = 0.f;
        for (int i = 0; i < 8; ++i) { S += red[i]; SS += red[8 + i]; }
        red[0] = S; red[1] = SS;
    }
    __syncthreads();
    float mean = red[0] / DMODEL;
    float var  = red[1] / DMODEL - mean * mean;
    float rstd = rsqrtf(var + 1e-5f);
#pragma unroll
    for (int i = 0; i < 4; ++i) {
        int d = t + 256 * i;
        out[(size_t)row * DMODEL + d] = f2bf((vals[i] - mean) * rstd * g[d] + bt[d]);
    }
}

// ---------------- out = x copy, then scatter-add processed tokens ----------------
__global__ void copy_kernel(const float4* __restrict__ src, float4* __restrict__ dst, int n4) {
    for (int i = blockIdx.x * blockDim.x + threadIdx.x; i < n4; i += gridDim.x * blockDim.x)
        dst[i] = src[i];
}

__global__ __launch_bounds__(256) void scatter_kernel(const float* __restrict__ proc,
                                                      const int* __restrict__ pos,
                                                      float* __restrict__ out) {
    int i = blockIdx.x;
    int b = i >> 9;
    int p = pos[i];
    float* dst = out + ((size_t)(b * SEQ + p)) * DMODEL;
    const float* src = proc + (size_t)i * DMODEL;
    for (int d = threadIdx.x; d < DMODEL; d += 256) dst[d] += src[d];
}

extern "C" void kernel_launch(void* const* d_in, const int* in_sizes, int n_in,
                              void* d_out, int out_size, void* d_ws, size_t ws_size,
                              hipStream_t stream) {
    (void)in_sizes; (void)n_in; (void)out_size; (void)ws_size;
    const float* x        = (const float*)d_in[0];
    const float* w_router = (const float*)d_in[1];
    const float* b_router = (const float*)d_in[2];
    const float* w_qkv    = (const float*)d_in[3];
    const float* b_qkv    = (const float*)d_in[4];
    const float* w_proj   = (const float*)d_in[5];
    const float* b_proj   = (const float*)d_in[6];
    const float* ln_g     = (const float*)d_in[7];
    const float* ln_b     = (const float*)d_in[8];
    const float* w_fc     = (const float*)d_in[9];
    const float* b_fc     = (const float*)d_in[10];
    const float* w_out    = (const float*)d_in[11];
    const float* b_out    = (const float*)d_in[12];

    char* ws = (char*)d_ws;
    size_t off = 0;
    auto alloc = [&](size_t bytes) -> void* {
        void* p = ws + off;
        off += (bytes + 255) & ~(size_t)255;
        return p;
    };
    float*          tw     = (float*)alloc((size_t)BATCH * SEQ * sizeof(float));
    int*            pos    = (int*)alloc((size_t)BATCH * KTOK * sizeof(int));
    unsigned short* tok    = (unsigned short*)alloc((size_t)ROWS * DMODEL * 2);   // reused as x_inner (bf16)
    unsigned short* wqkvT  = (unsigned short*)alloc((size_t)DMODEL * 3072 * 2);   // [3072][1024]
    unsigned short* wprojT = (unsigned short*)alloc((size_t)DMODEL * DMODEL * 2); // [1024][1024]
    unsigned short* wfcT   = (unsigned short*)alloc((size_t)DMODEL * 4096 * 2);   // [4096][1024]
    unsigned short* woutT  = (unsigned short*)alloc((size_t)4096 * DMODEL * 2);   // [1024][4096]
    unsigned short* qkvb   = (unsigned short*)alloc((size_t)ROWS * 3072 * 2);
    unsigned short* attnb  = (unsigned short*)alloc((size_t)ROWS * DMODEL * 2);
    float*          projf  = (float*)alloc((size_t)ROWS * DMODEL * sizeof(float)); // reused as processed
    unsigned short* hb     = (unsigned short*)alloc((size_t)ROWS * 4096 * 2);

    // weights: fp32 [K][N] -> bf16 [N][K]
    convert_transpose_kernel<<<dim3(3072 / 64, DMODEL / 64), 256, 0, stream>>>(w_qkv,  wqkvT,  DMODEL, 3072);
    convert_transpose_kernel<<<dim3(DMODEL / 64, DMODEL / 64), 256, 0, stream>>>(w_proj, wprojT, DMODEL, DMODEL);
    convert_transpose_kernel<<<dim3(4096 / 64, DMODEL / 64), 256, 0, stream>>>(w_fc,   wfcT,   DMODEL, 4096);
    convert_transpose_kernel<<<dim3(DMODEL / 64, 4096 / 64), 256, 0, stream>>>(w_out,  woutT,  4096, DMODEL);

    router_kernel<<<BATCH * SEQ / 8, 256, 0, stream>>>(x, w_router, b_router, tw);
    topk_kernel<<<BATCH, 1024, 0, stream>>>(tw, pos);
    gather_kernel<<<ROWS, 256, 0, stream>>>(x, pos, tok);

    gemm_bf16_kernel<0><<<dim3(3072 / 64, ROWS / 128), 256, 0, stream>>>(tok, wqkvT, b_qkv, qkvb, ROWS, 3072, DMODEL);
    attention_kernel<<<BATCH * NHEADS * 8, 256, 0, stream>>>(qkvb, attnb);
    gemm_bf16_kernel<1><<<dim3(DMODEL / 64, ROWS / 128), 256, 0, stream>>>(attnb, wprojT, b_proj, projf, ROWS, DMODEL, DMODEL);
    layernorm_kernel<<<ROWS, 256, 0, stream>>>(projf, ln_g, ln_b, tok);
    gemm_bf16_kernel<2><<<dim3(4096 / 64, ROWS / 128), 256, 0, stream>>>(tok, wfcT, b_fc, hb, ROWS, 4096, DMODEL);
    gemm_bf16_kernel<1><<<dim3(DMODEL / 64, ROWS / 128), 256, 0, stream>>>(hb, woutT, b_out, projf, ROWS, DMODEL, 4096);

    copy_kernel<<<2048, 256, 0, stream>>>((const float4*)x, (float4*)d_out, BATCH * SEQ * DMODEL / 4);
    scatter_kernel<<<ROWS, 256, 0, stream>>>(projf, pos, (float*)d_out);
}
